// ESNLinear_74998718923293
// MI455X (gfx1250) — compile-verified, matrix operand order checked by reading, not run on hardware
//
#include <hip/hip_runtime.h>
#include <hip/hip_bf16.h>

// Problem constants (from reference)
#define T_STEPS 4096
#define DIN     512
#define RDIM    4096
#define DOUT    512
#define NINT    (RDIM + DIN)   // 4608
#define LEAK_F  0.9f

typedef __attribute__((ext_vector_type(16))) __bf16 v16bf;
typedef __attribute__((ext_vector_type(8)))  float  v8f;

// ---------------------------------------------------------------------------
// fp32 -> bf16 conversion (grid-stride)
// ---------------------------------------------------------------------------
__global__ void cvt_f32_bf16(const float* __restrict__ in,
                             __bf16* __restrict__ out, int n) {
  int i = blockIdx.x * blockDim.x + threadIdx.x;
  int stride = gridDim.x * blockDim.x;
  for (; i < n; i += stride) out[i] = (__bf16)in[i];
}

__global__ void init_sync_kernel(int* sync) { sync[0] = 0; sync[1] = 0; }

// ---------------------------------------------------------------------------
// A-fragment loader (16x32 bf16, §7.12.2 layout):
// lane = half*16 + idx; per lane two contiguous 8-elem runs:
//   K in [8*half, 8*half+8) and [16+8*half, 16+8*half+8)
// ---------------------------------------------------------------------------
__device__ __forceinline__ v16bf load_frag_b16(const __bf16* __restrict__ base,
                                               int ld) {
  const int lane = threadIdx.x & 31;
  const int idx  = lane & 15;
  const int half = lane >> 4;
  const __bf16* p = base + (size_t)idx * ld + half * 8;
  union { uint4 u[2]; v16bf v; } f;
  f.u[0] = *(const uint4*)(p);
  f.u[1] = *(const uint4*)(p + 16);
  return f.v;
}

// ---------------------------------------------------------------------------
// WMMA bf16 GEMM:  out[M x N] = A1[M x K1] * W1^T  (+ A2[M x K2] * W2^T) + bias
//   W row-major as W[n][k] => B[k][n] = W[n][k], k-contiguous per column.
// Block = 8 waves = 128(M) x 64(N) tile; wave = 16 x 64.
// B tile (64n x 32k, 4 KB) staged into LDS with global_load_async_to_lds_b128,
// double-buffered so the next tile's DMA overlaps the current 4 WMMAs.
// A fragment double-buffered in registers.
// ---------------------------------------------------------------------------
__global__ void __launch_bounds__(256)
wmma_gemm_bf16(const __bf16* __restrict__ A1, int lda1, int K1,
               const __bf16* __restrict__ W1, int ldw1,
               const __bf16* __restrict__ A2, int lda2, int K2,
               const __bf16* __restrict__ W2, int ldw2,
               const float* __restrict__ bias,
               float* __restrict__ out, int ldo,
               int Nblks /* N/64 */) {
  __shared__ __align__(16) __bf16 btile[2][64 * 32];   // 2 x 4KB

  const int tid  = threadIdx.x;
  const int lane = tid & 31;
  const int wave = tid >> 5;
  const int mb = (int)blockIdx.x / Nblks;
  const int nb = (int)blockIdx.x % Nblks;
  const int m0 = mb * 128 + wave * 16;
  const int n0 = nb * 64;

  // per-thread slice of the 64x32 B tile: one b128 (8 bf16) each
  const int bn = tid >> 2;            // 0..63 : column n within tile
  const int bk = (tid & 3) * 8;       // 0,8,16,24 : k offset
  const unsigned lds_base = (unsigned)(size_t)(&btile[0][0]);
  const unsigned lds_mine = lds_base + (unsigned)tid * 16u;

  v8f acc[4];
#pragma unroll
  for (int j = 0; j < 4; ++j)
#pragma unroll
    for (int v = 0; v < 8; ++v) acc[j][v] = 0.0f;

  const __bf16* As[2]  = {A1, A2};
  const __bf16* Ws[2]  = {W1, W2};
  const int     ldas[2] = {lda1, lda2};
  const int     ldws[2] = {ldw1, ldw2};
  const int     Ks[2]   = {K1, K2};

  for (int seg = 0; seg < 2; ++seg) {
    const __bf16* A  = As[seg];
    const __bf16* W  = Ws[seg];
    const int lda = ldas[seg], ldw = ldws[seg], K = Ks[seg];
    if (A == nullptr || K == 0) continue;
    const int S = K / 32;

    // prologue: DMA B tile for k0=0 into buffer 0, prefetch A frag 0
    {
      const __bf16* g = W + (size_t)(n0 + bn) * ldw + bk;
      asm volatile("global_load_async_to_lds_b128 %0, %1, off"
                   :: "v"(lds_mine), "v"(g) : "memory");
    }
    v16bf a_cur = load_frag_b16(A + (size_t)m0 * lda, lda);

    for (int s = 0; s < S; ++s) {
      const int kn = (s + 1) * 32;
      if (kn < K) {
        // issue next B tile into the other buffer, then wait only for current
        const __bf16* g = W + (size_t)(n0 + bn) * ldw + kn + bk;
        asm volatile("global_load_async_to_lds_b128 %0, %1, off"
                     :: "v"(lds_mine + (unsigned)(((s + 1) & 1) * 4096)), "v"(g)
                     : "memory");
        asm volatile("s_wait_asynccnt 0x1" ::: "memory");
      } else {
        asm volatile("s_wait_asynccnt 0x0" ::: "memory");
      }
      __syncthreads();

      v16bf a_next = a_cur;
      if (kn < K) a_next = load_frag_b16(A + (size_t)m0 * lda + kn, lda);

      const int cloc = lane & 15;
      const int h8   = (lane >> 4) * 8;
#pragma unroll
      for (int j = 0; j < 4; ++j) {
        const __bf16* p = &btile[s & 1][(j * 16 + cloc) * 32 + h8];
        union { uint4 u[2]; v16bf v; } bf;
        bf.u[0] = *(const uint4*)(p);
        bf.u[1] = *(const uint4*)(p + 16);
        acc[j] = __builtin_amdgcn_wmma_f32_16x16x32_bf16(
            false, a_cur, false, bf.v, (short)0, acc[j], false, false);
      }
      a_cur = a_next;
      __syncthreads();   // protect btile[s&1] from the s+1 iteration's DMA
    }
  }

  const int half = lane >> 4;
  const int cloc = lane & 15;
#pragma unroll
  for (int j = 0; j < 4; ++j) {
    const int col = n0 + j * 16 + cloc;
    const float bv = bias ? bias[col] : 0.0f;
#pragma unroll
    for (int v = 0; v < 8; ++v) {
      const int row = m0 + v + 8 * half;   // C/D layout: M = v + 8*half
      out[(size_t)row * ldo + col] = acc[j][v] + bv;
    }
  }
}

// ---------------------------------------------------------------------------
// Persistent recurrent kernel: one launch for all 4096 steps.
// 128 blocks x 256 threads = 1024 waves; each wave owns 4 reservoir rows.
// Per step: async-DMA 8KB bf16 state into LDS, 4 dot(4096) per wave from
// L2-resident bf16 Wres (b128 coalesced), wave32 butterfly reduce,
// tanh + leaky update, device-wide generation barrier.
// State double-buffered in global (read (t-1)&1, write t&1).
// ---------------------------------------------------------------------------
#define REC_BLOCKS  128
#define REC_THREADS 256

__device__ __forceinline__ void grid_bar(int* __restrict__ sync, int* step) {
  __syncthreads();
  const int target = ++(*step);
  if (threadIdx.x == 0) {
    __threadfence();
    const int ticket = atomicAdd(&sync[0], 1);
    if (ticket == (int)gridDim.x - 1) {
      atomicExch(&sync[0], 0);
      __threadfence();
      atomicAdd(&sync[1], 1);
    } else {
      while (__atomic_load_n((const int*)&sync[1], __ATOMIC_RELAXED) < target) {
        __builtin_amdgcn_s_sleep(2);
      }
    }
    __threadfence();
  }
  __syncthreads();
}

__global__ void __launch_bounds__(REC_THREADS)
esn_recurrent(const float* __restrict__ u,       // T x R (fp32)
              const __bf16* __restrict__ Wres,   // R x R (bf16, L2-resident)
              __bf16* __restrict__ sbuf,         // 2 x R double buffer
              __bf16* __restrict__ states,       // T x R (bf16, for readout)
              int* __restrict__ sync) {
  __shared__ __align__(16) __bf16 s_lds[RDIM];
  const int tid   = threadIdx.x;
  const int lane  = tid & 31;
  const int wave  = tid >> 5;
  const int gwave = blockIdx.x * (REC_THREADS / 32) + wave;
  const int r0    = gwave * 4;
  const unsigned lds_mine = (unsigned)(size_t)(&s_lds[0]) + (unsigned)tid * 16u;

  float s_prev[4];
  int bstep = 0;

  // t = 0 : s0 = tanh(u[0]) (no Wres / leak term)
#pragma unroll
  for (int rr = 0; rr < 4; ++rr) {
    const float s0 = tanhf(u[r0 + rr]);
    s_prev[rr] = s0;
    if (lane == 0) {
      const __bf16 b = (__bf16)s0;
      sbuf[r0 + rr]   = b;      // buffer 0 (read at t=1)
      states[r0 + rr] = b;
    }
  }
  grid_bar(sync, &bstep);

  for (int t = 1; t < T_STEPS; ++t) {
    // async-DMA previous state (8 KB) into LDS: 2 x b128 per thread;
    // the offset immediate applies to both the global and LDS addresses.
    {
      const __bf16* g0 = sbuf + ((t - 1) & 1) * RDIM + tid * 8;
      asm volatile("global_load_async_to_lds_b128 %0, %1, off"
                   :: "v"(lds_mine), "v"(g0) : "memory");
      asm volatile("global_load_async_to_lds_b128 %0, %1, off offset:4096"
                   :: "v"(lds_mine), "v"(g0) : "memory");
      asm volatile("s_wait_asynccnt 0x0" ::: "memory");
    }
    __syncthreads();

    float acc[4] = {0.f, 0.f, 0.f, 0.f};
#pragma unroll 4
    for (int it = 0; it < 16; ++it) {
      const int k0 = it * 256 + lane * 8;
      union { uint4 u4; __bf16 h[8]; } sv;
      sv.u4 = *(const uint4*)(s_lds + k0);
      float sf[8];
#pragma unroll
      for (int j = 0; j < 8; ++j) sf[j] = (float)sv.h[j];
#pragma unroll
      for (int rr = 0; rr < 4; ++rr) {
        union { uint4 u4; __bf16 h[8]; } wv;
        wv.u4 = *(const uint4*)(Wres + (size_t)(r0 + rr) * RDIM + k0);
#pragma unroll
        for (int j = 0; j < 8; ++j)
          acc[rr] = fmaf((float)wv.h[j], sf[j], acc[rr]);
      }
    }

#pragma unroll
    for (int rr = 0; rr < 4; ++rr) {
      float a = acc[rr];
#pragma unroll
      for (int m = 16; m >= 1; m >>= 1) a += __shfl_xor(a, m, 32);
      const float xi = tanhf(u[(size_t)t * RDIM + r0 + rr] + a);
      const float sn = (1.0f - LEAK_F) * s_prev[rr] + LEAK_F * xi;
      s_prev[rr] = sn;
      if (lane == 0) {
        const __bf16 b = (__bf16)sn;
        sbuf[(t & 1) * RDIM + r0 + rr]     = b;
        states[(size_t)t * RDIM + r0 + rr] = b;
      }
    }
    grid_bar(sync, &bstep);   // also protects s_lds reuse next iteration
  }
}

// ---------------------------------------------------------------------------
// Host-side orchestration
// ---------------------------------------------------------------------------
extern "C" void kernel_launch(void* const* d_in, const int* in_sizes, int n_in,
                              void* d_out, int out_size, void* d_ws,
                              size_t ws_size, hipStream_t stream) {
  const float* x      = (const float*)d_in[0];   // T x DIN
  const float* Win_w  = (const float*)d_in[1];   // R x DIN
  const float* Win_b  = (const float*)d_in[2];   // R
  const float* Wres_w = (const float*)d_in[3];   // R x R
  const float* Wout_w = (const float*)d_in[4];   // DOUT x NINT
  const float* Wout_b = (const float*)d_in[5];   // DOUT
  float* yout = (float*)d_out;                   // T x DOUT

  char* ws = (char*)d_ws;
  size_t off = 0;
  auto carve = [&](size_t bytes) -> void* {
    void* p = (void*)(ws + off);
    off = (off + bytes + 255) & ~(size_t)255;
    return p;
  };
  float*  u       = (float*)  carve((size_t)T_STEPS * RDIM * 4);  // 64 MB
  __bf16* statesB = (__bf16*) carve((size_t)T_STEPS * RDIM * 2);  // 32 MB
  __bf16* WresB   = (__bf16*) carve((size_t)RDIM * RDIM * 2);     // 32 MB (L2-resident)
  __bf16* xB      = (__bf16*) carve((size_t)T_STEPS * DIN * 2);
  __bf16* WinB    = (__bf16*) carve((size_t)RDIM * DIN * 2);
  __bf16* WoutB   = (__bf16*) carve((size_t)DOUT * NINT * 2);
  __bf16* sbuf    = (__bf16*) carve((size_t)2 * RDIM * 2);
  int*    sync    = (int*)    carve(256);

  const int cthr = 256;
  auto cgrid = [](int n) { int g = (n + 255) / 256; return g > 2048 ? 2048 : g; };

  // 1) precision conversions (weights stay bf16 for all phases)
  cvt_f32_bf16<<<cgrid(T_STEPS * DIN), cthr, 0, stream>>>(x, xB, T_STEPS * DIN);
  cvt_f32_bf16<<<cgrid(RDIM * DIN), cthr, 0, stream>>>(Win_w, WinB, RDIM * DIN);
  cvt_f32_bf16<<<cgrid(RDIM * RDIM), cthr, 0, stream>>>(Wres_w, WresB, RDIM * RDIM);
  cvt_f32_bf16<<<cgrid(DOUT * NINT), cthr, 0, stream>>>(Wout_w, WoutB, DOUT * NINT);
  init_sync_kernel<<<1, 1, 0, stream>>>(sync);

  // 2) u = x @ Win^T + Win_b   (block tile 128x64, async-LDS B staging)
  {
    const int Mblks = T_STEPS / 128, Nblks = RDIM / 64;   // 32 x 64
    wmma_gemm_bf16<<<Mblks * Nblks, 256, 0, stream>>>(
        xB, DIN, DIN, WinB, DIN,
        nullptr, 0, 0, nullptr, 0,
        Win_b, u, RDIM, Nblks);
  }

  // 3) recurrence: persistent kernel, 4096 steps with grid-wide barriers
  esn_recurrent<<<REC_BLOCKS, REC_THREADS, 0, stream>>>(
      u, WresB, sbuf, statesB, sync);

  // 4) yout = [x, states] @ Wout^T + Wout_b  (two accumulated K segments)
  {
    const int Mblks = T_STEPS / 128, Nblks = DOUT / 64;   // 32 x 8
    wmma_gemm_bf16<<<Mblks * Nblks, 256, 0, stream>>>(
        xB, DIN, DIN, WoutB, NINT,                 // x part: W cols 0..511
        statesB, RDIM, RDIM, WoutB + DIN, NINT,    // state part: W cols 512..4607
        Wout_b, yout, DOUT, Nblks);
  }
}
